// HeteNet_84988812853490
// MI455X (gfx1250) — compile-verified, matrix operand order checked by reading, not run on hardware
//
#include <hip/hip_runtime.h>

// ---------------------------------------------------------------------------
// HeteNet distribute_compute for MI455X (gfx1250, wave32, WMMA).
// gather-by-expert -> grouped bf16 WMMA GEMMs (f32 accum) -> scatter.
// GEMM1: async-staged A tile in LDS (ASYNCcnt path), 32x64 output per wave.
// ---------------------------------------------------------------------------

typedef __attribute__((ext_vector_type(16))) __bf16 v16bf;
typedef __attribute__((ext_vector_type(8)))  float  v8f;

#define T_   64
#define N_   64
#define D_   512
#define H_   512
#define A_   32
#define E_   15
#define NTOK (T_ * N_)        // 4096 tokens
#define PCAP 4608             // padded-row capacity (4096 + 15*32, rounded)
#define MTILE 32              // M rows per tile (padded group granularity)
#define MAXTILES (PCAP / MTILE)  // 144

// ws layout (bytes)
#define OFF_NTILES 0
#define OFF_CURSOR 64
#define OFF_CNT    128
#define OFF_POFF   192
#define OFF_TILE_E 256
#define OFF_TILE_R 2048
#define OFF_PERM   4096                       // PCAP ints
#define OFF_XBF    32768                      // PCAP*D bf16
#define OFF_HBF    (OFF_XBF + PCAP * D_ * 2)  // PCAP*H bf16
#define OFF_W1T    (OFF_HBF + PCAP * H_ * 2)  // E*H*D bf16 (K-major)
#define OFF_W2T    (OFF_W1T + E_ * H_ * D_ * 2) // E*A*H bf16 (K-major)

__device__ __forceinline__ unsigned short f2bf(float f) {
  unsigned u = __float_as_uint(f);
  u += 0x7fffu + ((u >> 16) & 1u);   // round-to-nearest-even
  return (unsigned short)(u >> 16);
}

// ---------------------------------------------------------------------------
// 1) Build per-expert groups: counts, padded offsets (mult. of 32), token
//    permutation, and flat tile table.
// ---------------------------------------------------------------------------
__global__ __launch_bounds__(256) void k_build(const int* __restrict__ pick,
                                               int* ntiles, int* cursor, int* cnt,
                                               int* poff, int* tile_e, int* tile_r,
                                               int* perm) {
  __shared__ int scnt[E_];
  const int tid = threadIdx.x;
  if (tid < E_) scnt[tid] = 0;
  for (int i = tid; i < PCAP; i += blockDim.x) perm[i] = -1;
  __syncthreads();
  for (int t = tid; t < NTOK; t += blockDim.x) atomicAdd(&scnt[pick[t]], 1);
  __syncthreads();
  if (tid == 0) {
    int total = 0, tiles = 0;
    for (int e = 0; e < E_; ++e) {
      int c = scnt[e];
      cnt[e] = c; poff[e] = total; cursor[e] = 0;
      int pc = (c + MTILE - 1) & ~(MTILE - 1);
      for (int r = 0; r < pc; r += MTILE) { tile_e[tiles] = e; tile_r[tiles] = total + r; ++tiles; }
      total += pc;
    }
    *ntiles = tiles;
    __threadfence();
  }
  __syncthreads();
  for (int t = tid; t < NTOK; t += blockDim.x) {
    int e = pick[t];
    int p = atomicAdd(&cursor[e], 1);
    perm[poff[e] + p] = t;
  }
}

// ---------------------------------------------------------------------------
// 2) Convert weights to bf16, transposed to K-major so each B-fragment lane
//    reads 16 contiguous bf16 (32B).
// ---------------------------------------------------------------------------
__global__ __launch_bounds__(256) void k_w1t(const float* __restrict__ W1,
                                             unsigned short* __restrict__ W1T) {
  int idx = blockIdx.x * blockDim.x + threadIdx.x;
  if (idx >= E_ * D_ * H_) return;
  int h = idx % H_; int t = idx / H_; int d = t % D_; int e = t / D_;
  W1T[((size_t)(e * H_ + h)) * D_ + d] = f2bf(W1[idx]);
}

__global__ __launch_bounds__(256) void k_w2t(const float* __restrict__ W2,
                                             unsigned short* __restrict__ W2T) {
  int idx = blockIdx.x * blockDim.x + threadIdx.x;
  if (idx >= E_ * H_ * A_) return;
  int a = idx % A_; int t = idx / A_; int h = t % H_; int e = t / H_;
  W2T[((size_t)(e * A_ + a)) * H_ + h] = f2bf(W2[idx]);
}

// ---------------------------------------------------------------------------
// 3) Gather obs rows into expert-grouped padded bf16 matrix (pad rows = 0).
// ---------------------------------------------------------------------------
__global__ __launch_bounds__(256) void k_gather(const float* __restrict__ obs,
                                                const int* __restrict__ perm,
                                                unsigned short* __restrict__ Xbf) {
  int idx = blockIdx.x * blockDim.x + threadIdx.x;
  if (idx >= PCAP * D_) return;
  int d = idx % D_; int p = idx / D_;
  int tok = perm[p];
  float v = (tok >= 0) ? obs[(size_t)tok * D_ + d] : 0.0f;
  Xbf[idx] = f2bf(v);
}

// ---------------------------------------------------------------------------
// 4) GEMM1: Hid = relu(X @ W1[e] + b1[e]).
//    Block = 32 tokens x 256 hidden, 4 waves; wave = 32x64 (2x4 WMMA tiles).
//    A tile (32x512 bf16 = 32KB) staged once per block into LDS with
//    global_load_async_to_lds_b128. The LDS destination offset is taken from
//    the REAL address of Ash (ptrtoint) so the array escapes into the asm and
//    the "memory" clobber aliases it -> the fragment ds_loads cannot be
//    folded away (fixes the poison-A codegen seen previously).
// ---------------------------------------------------------------------------
__global__ __launch_bounds__(128) void k_gemm1(const unsigned short* __restrict__ Xbf,
                                               const unsigned short* __restrict__ W1T,
                                               const float* __restrict__ b1,
                                               const int* __restrict__ ntiles,
                                               const int* __restrict__ tile_e,
                                               const int* __restrict__ tile_r,
                                               unsigned short* __restrict__ Hbf) {
  __shared__ __align__(32) unsigned short Ash[MTILE * D_];  // 32 KB

  int tile = blockIdx.x;
  if (tile >= *ntiles) return;               // block-uniform: EXEC stays all-1s
  int e = tile_e[tile], r0 = tile_r[tile];
  int wave = threadIdx.x >> 5, lane = threadIdx.x & 31;
  int mrow = lane & 15, khalf = lane >> 4;

  // --- async stage A tile: 32 rows x 512 K bf16 -> LDS ---------------------
  {
    const unsigned short* gsrc = Xbf + (size_t)r0 * D_;
    // 32*512*2 B = 2048 x 16B transfers over 128 threads = 16 per thread
#pragma unroll
    for (int i = 0; i < 16; ++i) {
      int idx = threadIdx.x + i * 128;               // 16-byte unit index
      // generic LDS pointer: low 32 bits == AS3 byte offset (ISA 10.2),
      // exactly what VDST of GLOBAL_LOAD_ASYNC_TO_LDS wants.
      unsigned lds_off = (unsigned)(size_t)(&Ash[idx * 8]);
      const unsigned short* gp = gsrc + idx * 8;
      asm volatile("global_load_async_to_lds_b128 %0, %1, off"
                   :: "v"(lds_off), "v"(gp)
                   : "memory");
    }
    asm volatile("s_wait_asynccnt 0x0" ::: "memory");
    __syncthreads();
  }

  // --- fragment pointers ---------------------------------------------------
  int cb = blockIdx.y * 256 + wave * 64;     // wave's first hidden column
  const unsigned short* A0 = Ash + (size_t)(0 * 16 + mrow) * D_ + khalf * 16;
  const unsigned short* A1 = Ash + (size_t)(1 * 16 + mrow) * D_ + khalf * 16;
  const unsigned short* B0 = W1T + (size_t)(e * H_ + cb + 0 * 16 + mrow) * D_ + khalf * 16;
  const unsigned short* B1 = W1T + (size_t)(e * H_ + cb + 1 * 16 + mrow) * D_ + khalf * 16;
  const unsigned short* B2 = W1T + (size_t)(e * H_ + cb + 2 * 16 + mrow) * D_ + khalf * 16;
  const unsigned short* B3 = W1T + (size_t)(e * H_ + cb + 3 * 16 + mrow) * D_ + khalf * 16;

  v8f acc[2][4] = {};
#pragma unroll 4
  for (int k = 0; k < D_; k += 32) {
    v16bf a0 = *(const v16bf*)(A0 + k);
    v16bf a1 = *(const v16bf*)(A1 + k);
    v16bf b0 = *(const v16bf*)(B0 + k);
    v16bf b1v = *(const v16bf*)(B1 + k);
    v16bf b2v = *(const v16bf*)(B2 + k);
    v16bf b3v = *(const v16bf*)(B3 + k);
    acc[0][0] = __builtin_amdgcn_wmma_f32_16x16x32_bf16(false, a0, false, b0,  (short)0, acc[0][0], false, false);
    acc[0][1] = __builtin_amdgcn_wmma_f32_16x16x32_bf16(false, a0, false, b1v, (short)0, acc[0][1], false, false);
    acc[0][2] = __builtin_amdgcn_wmma_f32_16x16x32_bf16(false, a0, false, b2v, (short)0, acc[0][2], false, false);
    acc[0][3] = __builtin_amdgcn_wmma_f32_16x16x32_bf16(false, a0, false, b3v, (short)0, acc[0][3], false, false);
    acc[1][0] = __builtin_amdgcn_wmma_f32_16x16x32_bf16(false, a1, false, b0,  (short)0, acc[1][0], false, false);
    acc[1][1] = __builtin_amdgcn_wmma_f32_16x16x32_bf16(false, a1, false, b1v, (short)0, acc[1][1], false, false);
    acc[1][2] = __builtin_amdgcn_wmma_f32_16x16x32_bf16(false, a1, false, b2v, (short)0, acc[1][2], false, false);
    acc[1][3] = __builtin_amdgcn_wmma_f32_16x16x32_bf16(false, a1, false, b3v, (short)0, acc[1][3], false, false);
  }

  // --- epilogue: bias + relu + bf16 store ----------------------------------
#pragma unroll
  for (int ct = 0; ct < 4; ++ct) {
    int col = cb + ct * 16 + mrow;
    float bias = b1[e * H_ + col];
#pragma unroll
    for (int rt = 0; rt < 2; ++rt) {
#pragma unroll
      for (int j = 0; j < 8; ++j) {          // C/D: lane covers rows j + 8*khalf
        float v = acc[rt][ct][j] + bias;
        v = v > 0.0f ? v : 0.0f;
        Hbf[(size_t)(r0 + rt * 16 + j + 8 * khalf) * H_ + col] = f2bf(v);
      }
    }
  }
}

// ---------------------------------------------------------------------------
// 5) GEMM2: logits = Hid @ W2[e] + b2[e], scatter to out[token][action].
//    Block = 32 tokens x 32 actions, 2 waves; wave = 32x16 (B frag reused).
// ---------------------------------------------------------------------------
__global__ __launch_bounds__(64) void k_gemm2(const unsigned short* __restrict__ Hbf,
                                              const unsigned short* __restrict__ W2T,
                                              const float* __restrict__ b2,
                                              const int* __restrict__ ntiles,
                                              const int* __restrict__ tile_e,
                                              const int* __restrict__ tile_r,
                                              const int* __restrict__ perm,
                                              float* __restrict__ out) {
  int tile = blockIdx.x;
  if (tile >= *ntiles) return;
  int e = tile_e[tile], r0 = tile_r[tile];
  int wave = threadIdx.x >> 5, lane = threadIdx.x & 31;
  int mrow = lane & 15, khalf = lane >> 4;
  int col = wave * 16 + mrow;                // action column 0..31

  const unsigned short* A0 = Hbf + (size_t)(r0 + 0 * 16 + mrow) * H_ + khalf * 16;
  const unsigned short* A1 = Hbf + (size_t)(r0 + 1 * 16 + mrow) * H_ + khalf * 16;
  const unsigned short* Bc = W2T + (size_t)(e * A_ + col) * H_ + khalf * 16;

  v8f acc0 = {}, acc1 = {};
#pragma unroll 4
  for (int k = 0; k < H_; k += 32) {
    v16bf a0 = *(const v16bf*)(A0 + k);
    v16bf a1 = *(const v16bf*)(A1 + k);
    v16bf b  = *(const v16bf*)(Bc + k);
    acc0 = __builtin_amdgcn_wmma_f32_16x16x32_bf16(false, a0, false, b, (short)0, acc0, false, false);
    acc1 = __builtin_amdgcn_wmma_f32_16x16x32_bf16(false, a1, false, b, (short)0, acc1, false, false);
  }
  float bias = b2[e * A_ + col];
#pragma unroll
  for (int j = 0; j < 8; ++j) {
    int p0 = r0 + j + 8 * khalf;
    int p1 = p0 + 16;
    int t0 = perm[p0];
    int t1 = perm[p1];
    if (t0 >= 0) out[(size_t)t0 * A_ + col] = acc0[j] + bias;
    if (t1 >= 0) out[(size_t)t1 * A_ + col] = acc1[j] + bias;
  }
}

// ---------------------------------------------------------------------------
extern "C" void kernel_launch(void* const* d_in, const int* in_sizes, int n_in,
                              void* d_out, int out_size, void* d_ws, size_t ws_size,
                              hipStream_t stream) {
  (void)in_sizes; (void)n_in; (void)out_size; (void)ws_size;
  const float* obs = (const float*)d_in[0];
  const float* W1  = (const float*)d_in[1];
  const float* b1  = (const float*)d_in[2];
  const float* W2  = (const float*)d_in[3];
  const float* b2  = (const float*)d_in[4];
  const int*  pick = (const int*)d_in[5];
  float* out = (float*)d_out;

  char* ws = (char*)d_ws;
  int* ntiles = (int*)(ws + OFF_NTILES);
  int* cursor = (int*)(ws + OFF_CURSOR);
  int* cnt    = (int*)(ws + OFF_CNT);
  int* poff   = (int*)(ws + OFF_POFF);
  int* tile_e = (int*)(ws + OFF_TILE_E);
  int* tile_r = (int*)(ws + OFF_TILE_R);
  int* perm   = (int*)(ws + OFF_PERM);
  unsigned short* Xbf = (unsigned short*)(ws + OFF_XBF);
  unsigned short* Hbf = (unsigned short*)(ws + OFF_HBF);
  unsigned short* W1T = (unsigned short*)(ws + OFF_W1T);
  unsigned short* W2T = (unsigned short*)(ws + OFF_W2T);

  k_build<<<1, 256, 0, stream>>>(pick, ntiles, cursor, cnt, poff, tile_e, tile_r, perm);
  k_w1t<<<(E_ * D_ * H_ + 255) / 256, 256, 0, stream>>>(W1, W1T);
  k_w2t<<<(E_ * H_ * A_ + 255) / 256, 256, 0, stream>>>(W2, W2T);
  k_gather<<<(PCAP * D_ + 255) / 256, 256, 0, stream>>>(obs, perm, Xbf);
  k_gemm1<<<dim3(MAXTILES, H_ / 256), 128, 0, stream>>>(Xbf, W1T, b1, ntiles, tile_e, tile_r, Hbf);
  k_gemm2<<<MAXTILES, 64, 0, stream>>>(Hbf, W2T, b2, ntiles, tile_e, tile_r, perm, out);
}
